// MatchLSTMAttention_64072322121773
// MI455X (gfx1250) — matrix operand order repaired and checked by reading, not compiled, not run on hardware
//
#include <hip/hip_runtime.h>
#include <hip/hip_bf16.h>

// Problem dims (compile-time, from reference)
#define B_   128
#define T_   2048
#define DP_  256
#define DQ_  256
#define DO_  256

typedef __attribute__((ext_vector_type(16))) __bf16 v16bf;
typedef __attribute__((ext_vector_type(4)))  __bf16 v4bf;
typedef __attribute__((ext_vector_type(8)))  __bf16 v8bf_t;
typedef __attribute__((ext_vector_type(8)))  float  v8f;

// LDS row stride for staged W_q (bf16 elements): 256 + 16 pad -> 544 B/row,
// 136 dwords -> consecutive rows shift 8 banks: near-conflict-free b128 reads.
#define WQ_LDS_STRIDE 272

// Branchless tanh on the hardware transcendental path:
// tanh(x) = 1 - 2/(exp(2x)+1); clamp keeps exp finite, tanh(+-10) == +-1 in fp32.
__device__ __forceinline__ float fast_tanh(float x)
{
    float xc = fminf(fmaxf(x, -10.f), 10.f);
    float e  = __expf(2.f * xc);                     // v_exp_f32, no branches
    return 1.f - 2.f * __builtin_amdgcn_rcpf(e + 1.f); // v_rcp_f32
}

// ---------------------------------------------------------------------------
// Kernel 0: one-shot W_q fp32 -> bf16 (row-major), 65536 elements.
// ---------------------------------------------------------------------------
__global__ __launch_bounds__(256) void wq_to_bf16_kernel(
    const float* __restrict__ Wq, __bf16* __restrict__ Wqbf)
{
    const int i = blockIdx.x * 256 + threadIdx.x;    // 16384 float4 chunks
    float4 f = ((const float4*)Wq)[i];
    v4bf o;
    o[0] = (__bf16)f.x; o[1] = (__bf16)f.y; o[2] = (__bf16)f.z; o[3] = (__bf16)f.w;
    ((v4bf*)Wqbf)[i] = o;
}

// ---------------------------------------------------------------------------
// Kernel 1: P[b,n] = concat(input_p[b], h_tm1[b]) . W_p_r[n,:] + b_p_r[n] + b_q[n]
// ---------------------------------------------------------------------------
__global__ __launch_bounds__(256) void gpr_kernel(
    const float* __restrict__ input_p, const float* __restrict__ h_tm1,
    const float* __restrict__ Wpr, const float* __restrict__ bpr,
    const float* __restrict__ bq, float* __restrict__ P)
{
    const int b = blockIdx.x;    // B_
    const int n = threadIdx.x;   // DO_
    const float* wrow = Wpr + (size_t)n * (DP_ + DO_);
    const float* pb = input_p + (size_t)b * DP_;
    const float* hb = h_tm1   + (size_t)b * DO_;
    float acc = bpr[n] + bq[n];
#pragma unroll 4
    for (int k = 0; k < DP_; ++k) acc = fmaf(pb[k], wrow[k], acc);
#pragma unroll 4
    for (int k = 0; k < DO_; ++k) acc = fmaf(hb[k], wrow[DP_ + k], acc);
    P[b * DO_ + n] = acc;
}

// ---------------------------------------------------------------------------
// Kernel 2: fused  logit[b,t] = w . tanh(P[b,:] + W_q input_q[b,t,:]) + match_b
// v_wmma_f32_16x16x32_bf16; one wave per 16-row M-tile (16 | T so a tile
// stays in one batch). W_q (bf16) staged through LDS in 4 x 64-row slices
// shared by the block's 8 waves; B fragments come from ds_load_b128 with no
// per-tile conversion. A-panel converted fp32->bf16 once per tile (64 cvts).
// Epilogue uses branchless fast_tanh (no EXEC divergence anywhere).
// ---------------------------------------------------------------------------
__global__ __launch_bounds__(256) void fused_logits_kernel(
    const float* __restrict__ q,       // [B,T,DQ] fp32
    const __bf16* __restrict__ Wqbf,   // [DO,DQ] bf16 row-major
    const float* __restrict__ w,       // [DO]
    const float* __restrict__ matchb,  // [1]
    const float* __restrict__ P,       // [B,DO]  (includes b_q)
    float* __restrict__ logits)        // [B,T]
{
    __shared__ __bf16 sWq[64 * WQ_LDS_STRIDE];   // 34816 B

    const int tid    = threadIdx.x;
    const int lane   = tid & 31;
    const int wave   = tid >> 5;
    const int tile   = blockIdx.x * 8 + wave;   // 16-row tile index
    const int laneHi = lane >> 4;               // 0 or 1
    const int lan16  = lane & 15;

    const int b     = tile >> 7;                // T_/16 = 128 tiles per batch
    const int tbase = (tile & 127) << 4;        // first t of this tile

    // ---- Load A-panel once: row = lan16. bf16 A-fragment layout:
    // lanes 0-15 : K = kb..kb+7, kb+16..kb+23   (kb = ks*32)
    // lanes16-31 : same with kb = ks*32 + 8
    const float* qrow = q + ((size_t)tile * 16 + lan16) * DQ_;
    v16bf afrag[8];
#pragma unroll
    for (int ks = 0; ks < 8; ++ks) {
        const int kb = ks * 32 + laneHi * 8;
        float4 f0 = *(const float4*)(qrow + kb + 0);
        float4 f1 = *(const float4*)(qrow + kb + 4);
        float4 f2 = *(const float4*)(qrow + kb + 16);
        float4 f3 = *(const float4*)(qrow + kb + 20);
        v16bf a;
        a[0]=(__bf16)f0.x; a[1]=(__bf16)f0.y; a[2]=(__bf16)f0.z; a[3]=(__bf16)f0.w;
        a[4]=(__bf16)f1.x; a[5]=(__bf16)f1.y; a[6]=(__bf16)f1.z; a[7]=(__bf16)f1.w;
        a[8]=(__bf16)f2.x; a[9]=(__bf16)f2.y; a[10]=(__bf16)f2.z; a[11]=(__bf16)f2.w;
        a[12]=(__bf16)f3.x; a[13]=(__bf16)f3.y; a[14]=(__bf16)f3.z; a[15]=(__bf16)f3.w;
        afrag[ks] = a;
    }

    float part[8];
#pragma unroll
    for (int r = 0; r < 8; ++r) part[r] = 0.f;

    // ---- 4 stages: stage 64 rows of bf16 W_q into LDS, run 4 N-tiles each.
    for (int g = 0; g < 4; ++g) {
        // Cooperative copy: 64 rows x 16 chunks of 16 bf16 (32 B) = 1024 chunks.
        __syncthreads();   // previous stage's reads done before overwrite
#pragma unroll
        for (int it = 0; it < 4; ++it) {
            const int c   = it * 256 + tid;
            const int row = c >> 4;
            const int col = (c & 15) * 16;
            *(v16bf*)&sWq[row * WQ_LDS_STRIDE + col] =
                *(const v16bf*)(Wqbf + ((size_t)(g * 64 + row)) * DQ_ + col);
        }
        __syncthreads();

#pragma unroll
        for (int ntl = 0; ntl < 4; ++ntl) {
            const int nt   = g * 4 + ntl;
            const int n    = nt * 16 + lan16;
            const int lrow = ntl * 16 + lan16;     // row within staged slice
            v8f acc = {};
#pragma unroll
            for (int ks = 0; ks < 8; ++ks) {
                // B-fragment: lanes 0-15 K = kb..kb+7, kb+16..kb+23; lanes 16-31 kb+8.
                const __bf16* bp = &sWq[lrow * WQ_LDS_STRIDE + ks * 32 + laneHi * 8];
                const v8bf_t b0 = *(const v8bf_t*)(bp);
                const v8bf_t b1 = *(const v8bf_t*)(bp + 16);
                const v16bf bb = __builtin_shufflevector(b0, b1, 0,1,2,3,4,5,6,7,8,9,10,11,12,13,14,15);
                acc = __builtin_amdgcn_wmma_f32_16x16x32_bf16(
                          false, afrag[ks], false, bb, (short)0, acc, false, false);
            }
            // Epilogue: C layout -> VGPR r holds row (r + 8*laneHi), col n.
            const float pbn = P[b * DO_ + n];
            const float wn  = w[n];
#pragma unroll
            for (int r = 0; r < 8; ++r)
                part[r] = fmaf(wn, fast_tanh(acc[r] + pbn), part[r]);
        }
    }

    // ---- Reduce over the 16 lanes sharing a row (masks < 16 keep wave32
    // halves separate), then lane 0 / lane 16 write 8 rows each.
#pragma unroll
    for (int r = 0; r < 8; ++r) {
        float v = part[r];
        v += __shfl_xor(v, 8);
        v += __shfl_xor(v, 4);
        v += __shfl_xor(v, 2);
        v += __shfl_xor(v, 1);
        part[r] = v;
    }
    const float mb = matchb[0];
    if (lan16 == 0) {
        const int rowoff = laneHi * 8;
#pragma unroll
        for (int r = 0; r < 8; ++r)
            logits[b * T_ + tbase + rowoff + r] = part[r] + mb;
    }
}

// ---------------------------------------------------------------------------
// Kernel 3: masked softmax over T per batch (matches reference semantics:
// clip, multiply by mask, max includes masked zeros, e_x *= mask, +1e-6).
// exp argument is in [-30, 0] -> __expf is safe and branchless.
// ---------------------------------------------------------------------------
__global__ __launch_bounds__(256) void masked_softmax_kernel(
    const float* __restrict__ logits, const int* __restrict__ mask_q,
    float* __restrict__ alpha)
{
    const int b   = blockIdx.x;
    const int tid = threadIdx.x;
    __shared__ float red[8];

    // pass 1: max of clip(x)*m
    float mx = -3.0e38f;
    for (int t = tid; t < T_; t += 256) {
        float m = (float)mask_q[b * T_ + t];
        float x = fminf(fmaxf(logits[b * T_ + t], -15.f), 15.f) * m;
        mx = fmaxf(mx, x);
    }
    for (int off = 16; off >= 1; off >>= 1) mx = fmaxf(mx, __shfl_xor(mx, off));
    if ((tid & 31) == 0) red[tid >> 5] = mx;
    __syncthreads();
    const float bmax = fmaxf(fmaxf(fmaxf(red[0], red[1]), fmaxf(red[2], red[3])),
                             fmaxf(fmaxf(red[4], red[5]), fmaxf(red[6], red[7])));
    __syncthreads();

    // pass 2: sum of exp
    float sm = 0.f;
    for (int t = tid; t < T_; t += 256) {
        float m = (float)mask_q[b * T_ + t];
        float x = fminf(fmaxf(logits[b * T_ + t], -15.f), 15.f) * m;
        sm += __expf(x - bmax) * m;
    }
    for (int off = 16; off >= 1; off >>= 1) sm += __shfl_xor(sm, off);
    if ((tid & 31) == 0) red[tid >> 5] = sm;
    __syncthreads();
    const float bsum = red[0]+red[1]+red[2]+red[3]+red[4]+red[5]+red[6]+red[7];
    const float inv  = 1.f / (bsum + 1e-6f);

    // pass 3: normalize
    for (int t = tid; t < T_; t += 256) {
        float m = (float)mask_q[b * T_ + t];
        float x = fminf(fmaxf(logits[b * T_ + t], -15.f), 15.f) * m;
        alpha[b * T_ + t] = __expf(x - bmax) * m * inv;
    }
}

// ---------------------------------------------------------------------------
// Kernel 4a: partial context z over 16 T-chunks (coalesced 1 KB rows).
// ---------------------------------------------------------------------------
__global__ __launch_bounds__(256) void context_partial_kernel(
    const float* __restrict__ q, const float* __restrict__ alpha,
    float* __restrict__ zpart)   // [B,16,DQ]
{
    const int b = blockIdx.x;
    const int c = blockIdx.y;    // 16 chunks of 128 timesteps
    const int d = threadIdx.x;   // DQ_
    const float* qb = q + ((size_t)b * T_ + c * 128) * DQ_;
    const float* ab = alpha + b * T_ + c * 128;
    float acc = 0.f;
#pragma unroll 4
    for (int t = 0; t < 128; ++t)
        acc = fmaf(ab[t], qb[(size_t)t * DQ_ + d], acc);
    zpart[((size_t)b * 16 + c) * DQ_ + d] = acc;
}

// ---------------------------------------------------------------------------
// Kernel 4b: finalize out = concat([input_p, z], -1)
// ---------------------------------------------------------------------------
__global__ __launch_bounds__(256) void finalize_kernel(
    const float* __restrict__ input_p, const float* __restrict__ zpart,
    float* __restrict__ out)     // [B, DP+DQ]
{
    const int b = blockIdx.x;
    const int d = threadIdx.x;
    float acc = 0.f;
#pragma unroll
    for (int c = 0; c < 16; ++c)
        acc += zpart[((size_t)b * 16 + c) * DQ_ + d];
    out[(size_t)b * (DP_ + DQ_) + d]        = input_p[(size_t)b * DP_ + d];
    out[(size_t)b * (DP_ + DQ_) + DP_ + d]  = acc;
}

// ---------------------------------------------------------------------------
extern "C" void kernel_launch(void* const* d_in, const int* in_sizes, int n_in,
                              void* d_out, int out_size, void* d_ws, size_t ws_size,
                              hipStream_t stream)
{
    const float* input_p = (const float*)d_in[0];
    // d_in[1] = mask_p (unused by reference)
    const float* input_q = (const float*)d_in[2];
    const int*   mask_q  = (const int*)  d_in[3];
    const float* h_tm1   = (const float*)d_in[4];
    const float* W_p_r   = (const float*)d_in[5];
    const float* b_p_r   = (const float*)d_in[6];
    const float* W_q     = (const float*)d_in[7];
    const float* b_q     = (const float*)d_in[8];
    const float* w       = (const float*)d_in[9];
    const float* match_b = (const float*)d_in[10];
    float* out = (float*)d_out;

    // Workspace layout: Wqbf (bf16, 128 KB) | P | logits | alpha | zpart (fp32)
    __bf16* Wqbf  = (__bf16*)d_ws;
    float*  P     = (float*)((char*)d_ws + (size_t)DO_ * DQ_ * sizeof(__bf16));
    float* logits = P      + (size_t)B_ * DO_;
    float* alpha  = logits + (size_t)B_ * T_;
    float* zpart  = alpha  + (size_t)B_ * T_;

    wq_to_bf16_kernel<<<(DO_ * DQ_) / 4 / 256, 256, 0, stream>>>(W_q, Wqbf);

    gpr_kernel<<<B_, 256, 0, stream>>>(input_p, h_tm1, W_p_r, b_p_r, b_q, P);

    // (B*T/16) 16-row tiles, 8 waves (tiles) per 256-thread block
    const int mtiles = (B_ * T_) / 16;          // 16384
    fused_logits_kernel<<<mtiles / 8, 256, 0, stream>>>(
        input_q, Wqbf, w, match_b, P, logits);

    masked_softmax_kernel<<<B_, 256, 0, stream>>>(logits, mask_q, alpha);

    context_partial_kernel<<<dim3(B_, 16), 256, 0, stream>>>(input_q, alpha, zpart);

    finalize_kernel<<<B_, 256, 0, stream>>>(input_p, zpart, out);
}
